// DeepEMD_44238163149465
// MI455X (gfx1250) — matrix operand order repaired and checked by reading, not compiled, not run on hardware
//
#include <hip/hip_runtime.h>
#include <hip/hip_bf16.h>
#include <math.h>

// ---- types ------------------------------------------------------------
typedef __bf16 bf16_t;
typedef __attribute__((ext_vector_type(8)))  __bf16 v8bf;
typedef __attribute__((ext_vector_type(16))) __bf16 v16bf;
typedef __attribute__((ext_vector_type(8)))  float  v8f;

#define M_SAMPLES 128
#define C_DIM     640
#define HW        49
#define HW_PAD    64   // pad 49 -> 64 rows (4 WMMA row tiles), pad rows zeroed
#define TEMP_SCALE (12.5f / 49.0f)

__device__ __forceinline__ v16bf cat16(v8bf lo, v8bf hi) {
    return __builtin_shufflevector(lo, hi, 0,1,2,3,4,5,6,7,8,9,10,11,12,13,14,15);
}

__device__ __forceinline__ v8f wmma_bf16(v16bf a, v16bf b, v8f c) {
    // v_wmma_f32_16x16x32_bf16  D = A x B + C   (wave32)
    return __builtin_amdgcn_wmma_f32_16x16x32_bf16(
        /*neg_a=*/false, a, /*neg_b=*/false, b,
        /*c_mod=*/(short)0, c, /*reuse_a=*/false, /*reuse_b=*/false);
}

// A fragment: lane L holds row (base + L%16); elements 0..7 = K+(L<16?0:8),
// elements 8..15 = K+(L<16?16:24).  rowPtr points at row start (K-major).
__device__ __forceinline__ v16bf load_a_frag(const bf16_t* rowPtr, int k, int lane) {
    int o = (lane < 16) ? 0 : 8;
    v8bf lo = *(const v8bf*)(rowPtr + k + o);
    v8bf hi = *(const v8bf*)(rowPtr + k + o + 16);
    return cat16(lo, hi);
}

// B fragment: lane L holds column (base + L%16); elements 0..7 = K+(L<16?0:8),
// elements 8..15 = K+(L<16?16:24).
__device__ __forceinline__ v16bf load_b_frag(const bf16_t* colPtr, int k, int lane) {
    int o = (lane < 16) ? 0 : 8;
    v8bf lo = *(const v8bf*)(colPtr + k + o);
    v8bf hi = *(const v8bf*)(colPtr + k + o + 16);
    return cat16(lo, hi);
}

// ---- Phase A: per-sample preprocess -----------------------------------
// in:   [S][C][HW] f32
// raw:  [S][HW_PAD][C] bf16  (raw values, transposed, pad rows = 0)
// nrm:  [S][HW_PAD][C] bf16  (channel-centered, L2-normalized per pixel)
// meanHW: [S][C] bf16        (mean over HW of raw values)
__global__ __launch_bounds__(256)
void preprocess_kernel(const float* __restrict__ in,
                       bf16_t* __restrict__ raw,
                       bf16_t* __restrict__ nrm,
                       bf16_t* __restrict__ meanHW) {
    const int s   = blockIdx.x;
    const int tid = threadIdx.x;
    const float* x = in + (size_t)s * C_DIM * HW;

    __shared__ float s_cm[HW];   // per-pixel channel mean
    __shared__ float s_nm[HW];   // per-pixel L2 norm of centered vector

    if (tid < HW) {
        float acc = 0.f;
        for (int c = 0; c < C_DIM; ++c) acc += x[c * HW + tid];
        s_cm[tid] = acc / (float)C_DIM;
    }
    __syncthreads();
    if (tid < HW) {
        const float m = s_cm[tid];
        float acc = 0.f;
        for (int c = 0; c < C_DIM; ++c) {
            float v = x[c * HW + tid] - m;
            acc += v * v;
        }
        s_nm[tid] = fmaxf(sqrtf(acc), 1e-8f);
    }
    // spatial mean per channel (for the weight GEMMs)
    for (int c = tid; c < C_DIM; c += 256) {
        const float* row = x + (size_t)c * HW;
        float acc = 0.f;
        for (int p = 0; p < HW; ++p) acc += row[p];
        meanHW[(size_t)s * C_DIM + c] = (bf16_t)(acc / (float)HW);
    }
    __syncthreads();
    // transposed bf16 stores, zeroed pad rows
    bf16_t* rawS = raw + (size_t)s * HW_PAD * C_DIM;
    bf16_t* nrmS = nrm + (size_t)s * HW_PAD * C_DIM;
    for (int idx = tid; idx < HW_PAD * C_DIM; idx += 256) {
        const int p = idx / C_DIM;
        const int c = idx - p * C_DIM;
        float rv = 0.f, nv = 0.f;
        if (p < HW) {
            float v = x[(size_t)c * HW + p];
            rv = v;
            nv = (v - s_cm[p]) / s_nm[p];
        }
        rawS[idx] = (bf16_t)rv;
        nrmS[idx] = (bf16_t)nv;
    }
}

// ---- Phase B: weight GEMM ---------------------------------------------
// out[a][b][p] = relu( sum_c A[a][p][c] * Bm[b][c] ) + 0.001
// A:  [128][HW_PAD][C] bf16 (rows = a*64 + p), Bm: [128][C] bf16
// out: f32, index a*128*49 + b*49 + p
// grid = 512 blocks (one 16-row tile each), 8 waves = 8 column tiles
__global__ __launch_bounds__(256)
void weight_gemm_kernel(const bf16_t* __restrict__ A,
                        const bf16_t* __restrict__ Bm,
                        float* __restrict__ out) {
    const int tid  = threadIdx.x;
    const int lane = tid & 31;
    const int w    = tid >> 5;           // wave id = column tile
    const int rt   = blockIdx.x;         // row tile: rows rt*16 .. +15

    const int aRow = rt * 16 + (lane & 15);
    const int bCol = w * 16 + (lane & 15);
    const bf16_t* aPtr = A  + (size_t)aRow * C_DIM;
    const bf16_t* bPtr = Bm + (size_t)bCol * C_DIM;

    v8f acc = {};
    for (int k = 0; k < C_DIM; k += 32) {
        v16bf af = load_a_frag(aPtr, k, lane);
        v16bf bf = load_b_frag(bPtr, k, lane);
        acc = wmma_bf16(af, bf, acc);
    }

    // D layout: VGPR r -> row r (lanes 0-15) or r+8 (lanes 16-31); col = lane%16
    const int a     = rt >> 2;                    // sample index
    const int pbase = (rt & 3) * 16 + ((lane >= 16) ? 8 : 0);
    const int b     = w * 16 + (lane & 15);
    float* outA = out + (size_t)a * (128 * HW) + (size_t)b * HW;
#pragma unroll
    for (int r = 0; r < 8; ++r) {
        const int p = pbase + r;
        if (p < HW) outA[p] = fmaxf(acc[r], 0.f) + 0.001f;
    }
}

// ---- Phase C: EMD logits ----------------------------------------------
// Block = two (i,j) pairs: waves 0-3 -> j0, waves 4-7 -> j1 (shared Qn rows).
// Each wave owns a 2x2 block of 16x16 tiles: 2 A frags + 2 B frags per
// K-step feed 4 WMMAs (1 vmem load per WMMA).  sim tiles stay in registers,
// contracted with the 49x49 flow block, per-half LDS reduction.
__global__ __launch_bounds__(256)
void logits_kernel(const bf16_t* __restrict__ Qn,
                   const bf16_t* __restrict__ Pn,
                   const float* __restrict__ flow,
                   float* __restrict__ out) {
    const int tid  = threadIdx.x;
    const int lane = tid & 31;
    const int w    = tid >> 5;     // 0..7
    const int half = w >> 2;       // which j of the pair
    const int wl   = w & 3;        // wave within half

    const int i = blockIdx.y;
    const int j = blockIdx.x * 2 + half;

    const float* fblk = flow + ((size_t)(i * 128 + j)) * (HW * HW);

    // prefetch the 49x49 flow block (needed only after 20 WMMA K-steps)
    {
        const int t = tid & 127;                  // 128 threads per half
        if (t * 16 < HW * HW)         __builtin_prefetch(fblk + t * 16, 0, 0);
        if ((t + 128) * 16 < HW * HW) __builtin_prefetch(fblk + (t + 128) * 16, 0, 0);
    }

    const int rt0 = (wl >> 1) * 2;   // row tiles {rt0, rt0+1}
    const int ct0 = (wl & 1) * 2;    // col tiles {ct0, ct0+1}

    const bf16_t* a0Ptr = Qn + ((size_t)i * HW_PAD + rt0 * 16 + (lane & 15)) * C_DIM;
    const bf16_t* a1Ptr = a0Ptr + (size_t)16 * C_DIM;
    const bf16_t* b0Ptr = Pn + ((size_t)j * HW_PAD + ct0 * 16 + (lane & 15)) * C_DIM;
    const bf16_t* b1Ptr = b0Ptr + (size_t)16 * C_DIM;

    v8f c00 = {}, c01 = {}, c10 = {}, c11 = {};
    for (int k = 0; k < C_DIM; k += 32) {
        v16bf a0 = load_a_frag(a0Ptr, k, lane);
        v16bf a1 = load_a_frag(a1Ptr, k, lane);
        v16bf b0 = load_b_frag(b0Ptr, k, lane);
        v16bf b1 = load_b_frag(b1Ptr, k, lane);
        c00 = wmma_bf16(a0, b0, c00);
        c01 = wmma_bf16(a0, b1, c01);
        c10 = wmma_bf16(a1, b0, c10);
        c11 = wmma_bf16(a1, b1, c11);
    }

    // contract the four sim tiles with flow
    const int p0b = rt0 * 16 + ((lane >= 16) ? 8 : 0);
    const int q0  = ct0 * 16 + (lane & 15);
    const int q1  = q0 + 16;
    float s = 0.f, fs = 0.f;
#pragma unroll
    for (int r = 0; r < 8; ++r) {
        const int p0 = p0b + r;
        const int p1 = p0 + 16;
        if (p0 < HW) {
            if (q0 < HW) { float f = fblk[p0 * HW + q0]; fs += f; s += c00[r] * f; }
            if (q1 < HW) { float f = fblk[p0 * HW + q1]; fs += f; s += c01[r] * f; }
        }
        if (p1 < HW) {
            if (q0 < HW) { float f = fblk[p1 * HW + q0]; fs += f; s += c10[r] * f; }
            if (q1 < HW) { float f = fblk[p1 * HW + q1]; fs += f; s += c11[r] * f; }
        }
    }

    __shared__ float red[2][2];
    if (tid < 4) ((float*)red)[tid] = 0.f;
    __syncthreads();
    atomicAdd(&red[half][0], s);
    atomicAdd(&red[half][1], fs);
    __syncthreads();
    if ((tid & 127) == 0)
        out[i * 128 + j] = (red[half][1] - red[half][0]) * TEMP_SCALE;
}

// ---- host launcher -----------------------------------------------------
extern "C" void kernel_launch(void* const* d_in, const int* in_sizes, int n_in,
                              void* d_out, int out_size, void* d_ws, size_t ws_size,
                              hipStream_t stream) {
    (void)in_sizes; (void)n_in; (void)out_size; (void)ws_size;
    const float* query = (const float*)d_in[0];
    const float* proto = (const float*)d_in[1];
    const float* flow  = (const float*)d_in[2];
    float* out = (float*)d_out;

    // workspace carve-up (all chunks 16B-aligned)
    const size_t rawElems  = (size_t)M_SAMPLES * HW_PAD * C_DIM;   // 5,242,880
    const size_t meanElems = (size_t)M_SAMPLES * C_DIM;            // 81,920
    char* ws = (char*)d_ws;
    bf16_t* rawq  = (bf16_t*)ws; ws += rawElems  * sizeof(bf16_t);
    bf16_t* rawp  = (bf16_t*)ws; ws += rawElems  * sizeof(bf16_t);
    bf16_t* qn    = (bf16_t*)ws; ws += rawElems  * sizeof(bf16_t);
    bf16_t* pn    = (bf16_t*)ws; ws += rawElems  * sizeof(bf16_t);
    bf16_t* qmean = (bf16_t*)ws; ws += meanElems * sizeof(bf16_t);
    bf16_t* pmean = (bf16_t*)ws; ws += meanElems * sizeof(bf16_t);

    // output regions: [logits 16384][weight_1 802816][weight_2 802816]
    float* out_logits = out;
    float* out_w1 = out + 16384;
    float* out_w2 = out + 16384 + 128 * 128 * HW;

    preprocess_kernel<<<M_SAMPLES, 256, 0, stream>>>(query, rawq, qn, qmean);
    preprocess_kernel<<<M_SAMPLES, 256, 0, stream>>>(proto, rawp, pn, pmean);

    // weight_1[m,n,p] = relu(rawq[m,p,:] . pmean[n,:]) + 1e-3
    weight_gemm_kernel<<<512, 256, 0, stream>>>(rawq, pmean, out_w1);
    // weight_2[n,m,p] = relu(rawp[n,p,:] . qmean[m,:]) + 1e-3
    weight_gemm_kernel<<<512, 256, 0, stream>>>(rawp, qmean, out_w2);

    logits_kernel<<<dim3(64, 128), 256, 0, stream>>>(qn, pn, flow, out_logits);
}
